// GCN3D_78572131713481
// MI455X (gfx1250) — compile-verified, matrix operand order checked by reading, not run on hardware
//
#include <hip/hip_runtime.h>
#include <math.h>

// ---------------------------------------------------------------------------
// GCN3D for MI455X (gfx1250). GEMM-heavy parts use bf16 WMMA (f32 accum),
// irregular parts (knn/gather/softmax/bn) are f32 wave32 kernels.
// ---------------------------------------------------------------------------

typedef __attribute__((ext_vector_type(16))) __bf16 v16bf;
typedef __attribute__((ext_vector_type(8)))  float  v8f;

union FragBF {            // one WMMA 16-bit operand = two 16B LDS chunks
    v16bf v;
    uint4 q[2];
};
union PairBF {
    unsigned int u;
    __bf16 h[2];
};

// ------------------------------ WMMA GEMM ----------------------------------
// C[b] (MxN, row-major ldc) = A[b] (MxK, strides) * B[b] (KxN, strides) + bias
// optional relu and bf16 mirror of C.
// 128x64 macro tile, 8 waves as 4(M) x 2(N); each wave owns a 32x32 C tile:
// 2 A frags x 2 B frags -> 4 v_wmma_f32_16x16x32_bf16 per K-step.
// A staged row-major As[m][k]; B staged transposed Bsw[n][k] so both operands
// are fetched as two ds_load_b128 per fragment.
__global__ __launch_bounds__(256)
void gcn_gemm_bf16(const __bf16* __restrict__ Aptr, long asB, long asM, long asK,
                   const __bf16* __restrict__ Bptr, long bsB, long bsK, long bsN,
                   float* __restrict__ Cptr, long csB, int ldc,
                   int M, int N, int Kdim,
                   const float* __restrict__ bias, int relu,
                   __bf16* __restrict__ Cbf, long cbsB)
{
    __shared__ __align__(16) __bf16 As [128][40];  // 80B row stride (16B mult)
    __shared__ __align__(16) __bf16 Bsw[ 64][40];

    const int bz = blockIdx.z;
    const int m0 = blockIdx.y * 128;
    const int n0 = blockIdx.x * 64;
    const __bf16* Ab = Aptr + (long)bz * asB;
    const __bf16* Bb = Bptr + (long)bz * bsB;

    const int tid  = threadIdx.x;
    const int lane = tid & 31;
    const int wave = tid >> 5;
    const int wm   = wave & 3;   // 4 waves along M, 32 rows each
    const int wn   = wave >> 2;  // 2 waves along N, 32 cols each

    v8f acc00 = {}, acc01 = {}, acc10 = {}, acc11 = {};

    const int mrow  = lane & 15;
    const int khalf = (lane >> 4) * 8;
    const bool aPair = (asK == 1) && ((asM & 1) == 0);
    const bool bPairN = (bsN == 1) && ((bsK & 1) == 0);
    const bool bPairK = (bsK == 1) && ((bsN & 1) == 0);

    for (int k0 = 0; k0 < Kdim; k0 += 32) {
        const bool interior = (m0 + 128 <= M) && (n0 + 64 <= N) && (k0 + 32 <= Kdim);
        if (interior) {
            // ---- A tile 128x32 ----
            if (aPair) {
                for (int t = tid; t < 2048; t += 256) {        // 2048 bf16 pairs
                    int r = t >> 4, c = (t & 15) * 2;
                    unsigned int w = *(const unsigned int*)
                        (Ab + (long)(m0 + r) * asM + (k0 + c));
                    *(unsigned int*)&As[r][c] = w;
                }
            } else {
                for (int t = tid; t < 4096; t += 256) {
                    int r = t >> 5, c = t & 31;
                    As[r][c] = Ab[(long)(m0 + r) * asM + (long)(k0 + c) * asK];
                }
            }
            // ---- B tile 32x64 -> Bsw[n][k] ----
            if (bPairN) {
                for (int t = tid; t < 1024; t += 256) {
                    int gk = t >> 5, gn = (t & 31) * 2;
                    PairBF w;
                    w.u = *(const unsigned int*)(Bb + (long)(k0 + gk) * bsK + (n0 + gn));
                    Bsw[gn][gk]     = w.h[0];
                    Bsw[gn + 1][gk] = w.h[1];
                }
            } else if (bPairK) {
                for (int t = tid; t < 1024; t += 256) {
                    int gn = t >> 4, gk = (t & 15) * 2;
                    unsigned int w = *(const unsigned int*)
                        (Bb + (long)(n0 + gn) * bsN + (k0 + gk));
                    *(unsigned int*)&Bsw[gn][gk] = w;
                }
            } else {
                for (int t = tid; t < 2048; t += 256) {
                    int gk = t >> 6, gn = t & 63;
                    Bsw[gn][gk] = Bb[(long)(k0 + gk) * bsK + (long)(n0 + gn) * bsN];
                }
            }
        } else {
            for (int t = tid; t < 4096; t += 256) {
                int r = t >> 5, c = t & 31;
                int gm = m0 + r, gk = k0 + c;
                As[r][c] = (gm < M && gk < Kdim)
                             ? Ab[(long)gm * asM + (long)gk * asK] : (__bf16)0.0f;
            }
            for (int t = tid; t < 2048; t += 256) {
                int gk = k0 + (t >> 6), gn = n0 + (t & 63);
                Bsw[t & 63][t >> 6] = (gk < Kdim && gn < N)
                             ? Bb[(long)gk * bsK + (long)gn * bsN] : (__bf16)0.0f;
            }
        }
        __syncthreads();

        // fragments: contiguous 16B chunks at k-offsets khalf and khalf+16
        FragBF a0, a1, b0, b1;
        a0.q[0] = *(const uint4*)&As[wm * 32 +      mrow][khalf];
        a0.q[1] = *(const uint4*)&As[wm * 32 +      mrow][khalf + 16];
        a1.q[0] = *(const uint4*)&As[wm * 32 + 16 + mrow][khalf];
        a1.q[1] = *(const uint4*)&As[wm * 32 + 16 + mrow][khalf + 16];
        b0.q[0] = *(const uint4*)&Bsw[wn * 32 +      mrow][khalf];
        b0.q[1] = *(const uint4*)&Bsw[wn * 32 +      mrow][khalf + 16];
        b1.q[0] = *(const uint4*)&Bsw[wn * 32 + 16 + mrow][khalf];
        b1.q[1] = *(const uint4*)&Bsw[wn * 32 + 16 + mrow][khalf + 16];

        acc00 = __builtin_amdgcn_wmma_f32_16x16x32_bf16(
                    false, a0.v, false, b0.v, (short)0, acc00, false, false);
        acc01 = __builtin_amdgcn_wmma_f32_16x16x32_bf16(
                    false, a0.v, false, b1.v, (short)0, acc01, false, false);
        acc10 = __builtin_amdgcn_wmma_f32_16x16x32_bf16(
                    false, a1.v, false, b0.v, (short)0, acc10, false, false);
        acc11 = __builtin_amdgcn_wmma_f32_16x16x32_bf16(
                    false, a1.v, false, b1.v, (short)0, acc11, false, false);
        __syncthreads();
    }

    // epilogue: C frag layout — VGPR r: lanes0-15 -> M=r, lanes16-31 -> M=r+8
    #pragma unroll
    for (int t = 0; t < 2; ++t) {
        int ncol = n0 + wn * 32 + t * 16 + (lane & 15);
        if (ncol >= N) continue;
        float bb = (bias != nullptr) ? bias[ncol] : 0.0f;
        #pragma unroll
        for (int s = 0; s < 2; ++s) {
            const v8f& acc = (s == 0) ? (t == 0 ? acc00 : acc01)
                                      : (t == 0 ? acc10 : acc11);
            #pragma unroll
            for (int r = 0; r < 8; ++r) {
                int gm = m0 + wm * 32 + s * 16 + r + (lane >> 4) * 8;
                if (gm < M) {
                    float vv = acc[r] + bb;
                    if (relu) vv = fmaxf(vv, 0.0f);
                    Cptr[(long)bz * csB + (long)gm * ldc + ncol] = vv;
                    if (Cbf) Cbf[(long)bz * cbsB + (long)gm * ldc + ncol] = (__bf16)vv;
                }
            }
        }
    }
}

// ---------------------------- elementwise utils ----------------------------
__global__ void gcn_cvt_bf16(const float* __restrict__ x, __bf16* __restrict__ y, long n)
{
    long i = (long)blockIdx.x * blockDim.x + threadIdx.x;
    if (i < n) y[i] = (__bf16)x[i];
}

__global__ void gcn_copy_cvt(const float* __restrict__ x, float* __restrict__ y,
                             __bf16* __restrict__ yb, long n)
{
    long i = (long)blockIdx.x * blockDim.x + threadIdx.x;
    if (i < n) { float v = x[i]; y[i] = v; yb[i] = (__bf16)v; }
}

__global__ void gcn_sub_bf16(const float* __restrict__ a, const float* __restrict__ b,
                             __bf16* __restrict__ o, long n)
{
    long i = (long)blockIdx.x * blockDim.x + threadIdx.x;
    if (i < n) o[i] = (__bf16)(a[i] - b[i]);
}

// vertices (B,3,N) -> v (B,N,3)
__global__ void gcn_transpose_v(const float* __restrict__ in, float* __restrict__ out,
                                int n, long total)
{
    long i = (long)blockIdx.x * blockDim.x + threadIdx.x;
    if (i >= total) return;
    long b  = i / (3L * n);
    long r  = i % (3L * n);
    int  ch = (int)(r / n);
    int  p  = (int)(r % n);
    out[(b * n + p) * 3 + ch] = in[i];
}

// normalize columns of a (3,m) direction matrix
__global__ void gcn_norm_dirs(const float* __restrict__ d, float* __restrict__ o, int m)
{
    int c = blockIdx.x * blockDim.x + threadIdx.x;
    if (c >= m) return;
    float a = d[c], b = d[m + c], e = d[2*m + c];
    float inv = 1.0f / fmaxf(sqrtf(a*a + b*b + e*e), 1e-12f);
    o[c] = a*inv; o[m + c] = b*inv; o[2*m + c] = e*inv;
}

// ------------------------------ KNN (wave32) -------------------------------
__global__ __launch_bounds__(32)
void gcn_knn(const float* __restrict__ pts, long vbs, int npts,
             int* __restrict__ outIdx, int kout)
{
    __shared__ float d2s[2048];
    int b = blockIdx.y, i = blockIdx.x, lane = threadIdx.x;
    const float* vb = pts + (long)b * vbs;
    float qx = vb[(long)i*3], qy = vb[(long)i*3+1], qz = vb[(long)i*3+2];
    for (int j = lane; j < npts; j += 32) {
        float dx = vb[(long)j*3]-qx, dy = vb[(long)j*3+1]-qy, dz = vb[(long)j*3+2]-qz;
        d2s[j] = dx*dx + dy*dy + dz*dz;
    }
    __syncthreads();
    float lastD = -3.4e38f; int lastI = -1;
    int* orow = outIdx + ((long)b * gridDim.x + i) * kout;
    for (int r = 0; r <= kout; ++r) {
        float bd = 3.4e38f; int bi = 0x7fffffff;
        for (int j = lane; j < npts; j += 32) {
            float d = d2s[j];
            bool cand = (d > lastD) || (d == lastD && j > lastI);
            if (cand && (d < bd || (d == bd && j < bi))) { bd = d; bi = j; }
        }
        for (int off = 16; off; off >>= 1) {
            float od = __shfl_xor(bd, off, 32);
            int   oi = __shfl_xor(bi, off, 32);
            if (od < bd || (od == bd && oi < bi)) { bd = od; bi = oi; }
        }
        if (r > 0 && lane == 0) orow[r-1] = bi;
        lastD = bd; lastI = bi;
    }
}

// pool: 4-NN max-gather of features for the first pn points
__global__ __launch_bounds__(32)
void gcn_pool(const float* __restrict__ pts, long vbs, int npts,
              const float* __restrict__ fm, int C, int pn,
              float* __restrict__ outf, __bf16* __restrict__ outb)
{
    __shared__ float d2s[2048];
    __shared__ int nbs[4];
    int b = blockIdx.y, i = blockIdx.x, lane = threadIdx.x;
    const float* vb = pts + (long)b * vbs;
    float qx = vb[(long)i*3], qy = vb[(long)i*3+1], qz = vb[(long)i*3+2];
    for (int j = lane; j < npts; j += 32) {
        float dx = vb[(long)j*3]-qx, dy = vb[(long)j*3+1]-qy, dz = vb[(long)j*3+2]-qz;
        d2s[j] = dx*dx + dy*dy + dz*dz;
    }
    __syncthreads();
    float lastD = -3.4e38f; int lastI = -1;
    for (int r = 0; r <= 4; ++r) {
        float bd = 3.4e38f; int bi = 0x7fffffff;
        for (int j = lane; j < npts; j += 32) {
            float d = d2s[j];
            bool cand = (d > lastD) || (d == lastD && j > lastI);
            if (cand && (d < bd || (d == bd && j < bi))) { bd = d; bi = j; }
        }
        for (int off = 16; off; off >>= 1) {
            float od = __shfl_xor(bd, off, 32);
            int   oi = __shfl_xor(bi, off, 32);
            if (od < bd || (od == bd && oi < bi)) { bd = od; bi = oi; }
        }
        if (r > 0 && lane == 0) nbs[r-1] = bi;
        lastD = bd; lastI = bi;
    }
    __syncthreads();
    for (int c = lane; c < C; c += 32) {
        float mx = -3.4e38f;
        #pragma unroll
        for (int t = 0; t < 4; ++t)
            mx = fmaxf(mx, fm[((long)b*npts + nbs[t]) * C + c]);
        long o = ((long)b*pn + i) * C + c;
        outf[o] = mx; outb[o] = (__bf16)mx;
    }
}

__global__ __launch_bounds__(32)
void gcn_nearest(const float* __restrict__ v, long vbs, int ns,
                 int* __restrict__ out, int nt)
{
    int b = blockIdx.y, i = blockIdx.x, lane = threadIdx.x;
    const float* vb = v + (long)b * vbs;
    float qx = vb[(long)i*3], qy = vb[(long)i*3+1], qz = vb[(long)i*3+2];
    float bd = 3.4e38f; int bi = 0x7fffffff;
    for (int j = lane; j < ns; j += 32) {
        float dx = vb[(long)j*3]-qx, dy = vb[(long)j*3+1]-qy, dz = vb[(long)j*3+2]-qz;
        float d = dx*dx + dy*dy + dz*dz;
        if (d < bd || (d == bd && j < bi)) { bd = d; bi = j; }
    }
    for (int off = 16; off; off >>= 1) {
        float od = __shfl_xor(bd, off, 32);
        int   oi = __shfl_xor(bi, off, 32);
        if (od < bd || (od == bd && oi < bi)) { bd = od; bi = oi; }
    }
    if (lane == 0) out[(long)b*nt + i] = bi;
}

// ------------------------- direction conv kernels --------------------------
__global__ __launch_bounds__(128)
void gcn_conv_surface(const int* __restrict__ idx, const float* __restrict__ v, long vbs,
                      const float* __restrict__ dirs, float* __restrict__ out,
                      int n, int m)
{
    __shared__ float nd[50][3];
    int b = blockIdx.y, i = blockIdx.x, tid = threadIdx.x;
    const float* vb  = v + (long)b * vbs;
    const int*   row = idx + ((long)b*n + i) * 50;
    float qx = vb[(long)i*3], qy = vb[(long)i*3+1], qz = vb[(long)i*3+2];
    if (tid < 50) {
        int nb = row[tid];
        float dx = vb[(long)nb*3]-qx, dy = vb[(long)nb*3+1]-qy, dz = vb[(long)nb*3+2]-qz;
        float inv = 1.0f / fmaxf(sqrtf(dx*dx + dy*dy + dz*dz), 1e-12f);
        nd[tid][0] = dx*inv; nd[tid][1] = dy*inv; nd[tid][2] = dz*inv;
    }
    __syncthreads();
    for (int c = tid; c < m; c += 128) {
        float d0 = dirs[c], d1 = dirs[m + c], d2 = dirs[2*m + c];
        float mx = -3.4e38f;
        for (int j = 0; j < 50; ++j)
            mx = fmaxf(mx, nd[j][0]*d0 + nd[j][1]*d1 + nd[j][2]*d2);
        out[((long)b*n + i)*m + c] = fmaxf(mx, 0.0f);
    }
}

__global__ __launch_bounds__(128)
void gcn_conv_reduce(const int* __restrict__ idx, const float* __restrict__ v, long vbs,
                     const float* __restrict__ fout, const float* __restrict__ dirs,
                     float* __restrict__ out, int n, int cout)
{
    __shared__ float nd[50][3];
    __shared__ int   nbs[50];
    int b = blockIdx.y, i = blockIdx.x, tid = threadIdx.x;
    const float* vb  = v + (long)b * vbs;
    const int*   row = idx + ((long)b*n + i) * 50;
    float qx = vb[(long)i*3], qy = vb[(long)i*3+1], qz = vb[(long)i*3+2];
    if (tid < 50) {
        int nb = row[tid];
        nbs[tid] = nb;
        float dx = vb[(long)nb*3]-qx, dy = vb[(long)nb*3+1]-qy, dz = vb[(long)nb*3+2]-qz;
        float inv = 1.0f / fmaxf(sqrtf(dx*dx + dy*dy + dz*dz), 1e-12f);
        nd[tid][0] = dx*inv; nd[tid][1] = dy*inv; nd[tid][2] = dz*inv;
    }
    __syncthreads();
    const int  ld   = 2 * cout;
    const long base = (long)b*n + i;
    for (int c = tid; c < cout; c += 128) {
        float d0 = dirs[c], d1 = dirs[cout + c], d2 = dirs[2*cout + c];
        float mx = -3.4e38f;
        for (int j = 0; j < 50; ++j) {
            float th = fmaxf(nd[j][0]*d0 + nd[j][1]*d1 + nd[j][2]*d2, 0.0f);
            float fs = fout[((long)b*n + nbs[j])*ld + cout + c];
            mx = fmaxf(mx, th * fs);
        }
        out[base*cout + c] = fout[base*ld + c] + mx;
    }
}

// --------------------------- batchnorm / softmax ---------------------------
__global__ __launch_bounds__(256)
void gcn_bn_stats(const float* __restrict__ x, int M, int C,
                  float* __restrict__ mean, float* __restrict__ var)
{
    __shared__ float ssum[256], ssq[256];
    int c = blockIdx.x, tid = threadIdx.x;
    float s = 0.0f, sq = 0.0f;
    for (int r = tid; r < M; r += 256) {
        float v = x[(long)r*C + c];
        s += v; sq += v*v;
    }
    ssum[tid] = s; ssq[tid] = sq;
    __syncthreads();
    for (int off = 128; off; off >>= 1) {
        if (tid < off) { ssum[tid] += ssum[tid+off]; ssq[tid] += ssq[tid+off]; }
        __syncthreads();
    }
    if (tid == 0) {
        float m = ssum[0] / (float)M;
        mean[c] = m;
        var[c]  = ssq[0] / (float)M - m*m;
    }
}

__global__ void gcn_bn_apply(const float* __restrict__ x, const float* __restrict__ mean,
                             const float* __restrict__ var, const float* __restrict__ g,
                             const float* __restrict__ be, const float* __restrict__ res,
                             float* __restrict__ out, __bf16* __restrict__ outb,
                             long total, int C, int relu)
{
    long i = (long)blockIdx.x * blockDim.x + threadIdx.x;
    if (i >= total) return;
    int c = (int)(i % C);
    float v = (x[i] - mean[c]) * rsqrtf(var[c] + 1e-5f) * g[c] + be[c];
    if (relu) v = fmaxf(v, 0.0f);
    if (res)  v += res[i];
    out[i] = v;
    if (outb) outb[i] = (__bf16)v;
}

__global__ __launch_bounds__(256)
void gcn_softmax_rows(float* __restrict__ x, int L)
{
    __shared__ float red[256];
    long row = blockIdx.x;
    float* p = x + row * (long)L;
    int tid = threadIdx.x;
    float m = -3.4e38f;
    for (int i = tid; i < L; i += 256) m = fmaxf(m, p[i]);
    red[tid] = m; __syncthreads();
    for (int off = 128; off; off >>= 1) {
        if (tid < off) red[tid] = fmaxf(red[tid], red[tid+off]);
        __syncthreads();
    }
    m = red[0]; __syncthreads();
    float s = 0.0f;
    for (int i = tid; i < L; i += 256) { float e = expf(p[i] - m); p[i] = e; s += e; }
    red[tid] = s; __syncthreads();
    for (int off = 128; off; off >>= 1) {
        if (tid < off) red[tid] += red[tid+off];
        __syncthreads();
    }
    float inv = 1.0f / red[0];
    for (int i = tid; i < L; i += 256) p[i] *= inv;
}

__global__ void gcn_colsum(const float* __restrict__ attn, float* __restrict__ cs, int n)
{
    int b = blockIdx.y;
    int c = blockIdx.x * blockDim.x + threadIdx.x;
    if (c >= n) return;
    const float* p = attn + (long)b*n*n + c;
    float s = 0.0f;
    for (int r = 0; r < n; ++r) s += p[(long)r*n];
    cs[(long)b*n + c] = s;
}

__global__ void gcn_attn_scale_bf(const float* __restrict__ attn,
                                  const float* __restrict__ cs,
                                  __bf16* __restrict__ out, long total, int n)
{
    long i = (long)blockIdx.x * blockDim.x + threadIdx.x;
    if (i >= total) return;
    int  col = (int)(i % n);
    long b   = i / ((long)n * n);
    out[i] = (__bf16)(attn[i] / (1e-9f + cs[b*n + col]));
}

__global__ void gcn_rowmax(const float* __restrict__ x, int nr, int C,
                           float* __restrict__ out)
{
    int b = blockIdx.y;
    int c = blockIdx.x * blockDim.x + threadIdx.x;
    if (c >= C) return;
    float m = -3.4e38f;
    for (int r = 0; r < nr; ++r) m = fmaxf(m, x[((long)b*nr + r)*C + c]);
    out[(long)b*C + c] = m;
}

// fuse[b,i,:] = [fm0|fm1|fm2[ni1]|fm3[ni1]|fm4[ni2]|f_glob|onehot] as bf16
__global__ __launch_bounds__(256)
void gcn_fuse(const float* __restrict__ fm0, const float* __restrict__ fm1,
              const float* __restrict__ fm2, const float* __restrict__ fm3,
              const float* __restrict__ fm4, const float* __restrict__ fglob,
              const float* __restrict__ onehot, const int* __restrict__ ni1,
              const int* __restrict__ ni2, __bf16* __restrict__ out, int n)
{
    int b = blockIdx.y, i = blockIdx.x, tid = threadIdx.x;
    int i1 = ni1[(long)b*n + i];
    int i2 = ni2[(long)b*n + i];
    __bf16* row = out + ((long)b*n + i) * 1808;
    for (int c = tid; c < 1808; c += 256) {
        float v;
        if      (c <  128) v = fm0[((long)b*n   + i )*128 + c];
        else if (c <  256) v = fm1[((long)b*n   + i )*128 + (c - 128)];
        else if (c <  512) v = fm2[((long)b*512 + i1)*256 + (c - 256)];
        else if (c <  768) v = fm3[((long)b*512 + i1)*256 + (c - 512)];
        else if (c < 1280) v = fm4[((long)b*128 + i2)*512 + (c - 768)];
        else if (c < 1792) v = fglob[(long)b*512 + (c - 1280)];
        else               v = onehot[(long)b*16 + (c - 1792)];
        row[c] = (__bf16)v;
    }
}

__global__ __launch_bounds__(64)
void gcn_logsoftmax(const float* __restrict__ x, float* __restrict__ out, int L)
{
    __shared__ float red[64];
    long row = blockIdx.x;
    const float* p = x + row * (long)L;
    float* o = out + row * (long)L;
    int tid = threadIdx.x;
    float m = -3.4e38f;
    for (int i = tid; i < L; i += 64) m = fmaxf(m, p[i]);
    red[tid] = m; __syncthreads();
    for (int off = 32; off; off >>= 1) {
        if (tid < off) red[tid] = fmaxf(red[tid], red[tid+off]);
        __syncthreads();
    }
    m = red[0]; __syncthreads();
    float s = 0.0f;
    for (int i = tid; i < L; i += 64) s += expf(p[i] - m);
    red[tid] = s; __syncthreads();
    for (int off = 32; off; off >>= 1) {
        if (tid < off) red[tid] += red[tid+off];
        __syncthreads();
    }
    float lse = m + logf(red[0]);
    for (int i = tid; i < L; i += 64) o[i] = p[i] - lse;
}

// ------------------------------ host helpers -------------------------------
struct Bump {
    char*  base;
    size_t off;
    void* take(size_t bytes) {
        void* p = base + off;
        off = (off + bytes + 255) & ~(size_t)255;
        return p;
    }
};

static inline void gemm(hipStream_t s,
                        const __bf16* A, long asB, long asM, long asK,
                        const __bf16* B, long bsB, long bsK, long bsN,
                        float* C, long csB, int ldc, int M, int N, int K,
                        const float* bias, int relu, __bf16* Cbf, long cbsB,
                        int batch)
{
    dim3 g((N + 63) / 64, (M + 127) / 128, batch);
    gcn_gemm_bf16<<<g, 256, 0, s>>>(A, asB, asM, asK, B, bsB, bsK, bsN,
                                    C, csB, ldc, M, N, K, bias, relu, Cbf, cbsB);
}

static inline long ceil256(long n) { return (n + 255) / 256; }

// self-attention block, updates x (f32) and xb (bf16) in place
static void run_sa(hipStream_t s, Bump scr, float* x, __bf16* xb, int n, int c,
                   const __bf16* qkb, const __bf16* vwb, const float* vb_bias,
                   const __bf16* twb, const float* tb_bias,
                   const float* g, const float* be)
{
    const int  B  = 8;
    const int  cq = c / 4;
    const long nn = (long)n * n;
    float*  qf   = (float*) scr.take((size_t)B*n*cq*4);
    __bf16* qb   = (__bf16*)scr.take((size_t)B*n*cq*2);
    float*  attn = (float*) scr.take((size_t)B*nn*4);
    __bf16* attb = (__bf16*)scr.take((size_t)B*nn*2);
    float*  cs   = (float*) scr.take((size_t)B*n*4);
    float*  vf   = (float*) scr.take((size_t)B*n*c*4);
    __bf16* vbf  = (__bf16*)scr.take((size_t)B*n*c*2);
    float*  xr   = (float*) scr.take((size_t)B*n*c*4);
    __bf16* df   = (__bf16*)scr.take((size_t)B*n*c*2);
    float*  pre  = (float*) scr.take((size_t)B*n*c*4);
    float*  mv   = (float*) scr.take((size_t)c*8);

    // q = x @ qk^T
    gemm(s, xb, 0, c, 1, qkb, 0, 1, c, qf, 0, cq, B*n, cq, c,
         nullptr, 0, qb, 0, 1);
    // logits = q @ q^T (per batch)
    gemm(s, qb, (long)n*cq, cq, 1, qb, (long)n*cq, 1, cq, attn, nn, n,
         n, n, cq, nullptr, 0, nullptr, 0, B);
    gcn_softmax_rows<<<B*n, 256, 0, s>>>(attn, n);
    { dim3 g1((n + 255)/256, B); gcn_colsum<<<g1, 256, 0, s>>>(attn, cs, n); }
    { long tot = B*nn;
      gcn_attn_scale_bf<<<ceil256(tot), 256, 0, s>>>(attn, cs, attb, tot, n); }
    // v = x @ vw^T + vb
    gemm(s, xb, 0, c, 1, vwb, 0, 1, c, vf, 0, c, B*n, c, c,
         vb_bias, 0, vbf, 0, 1);
    // x_r = attn^T @ v (per batch)
    gemm(s, attb, nn, 1, n, vbf, (long)n*c, c, 1, xr, (long)n*c, c,
         n, c, n, nullptr, 0, nullptr, 0, B);
    { long tot = (long)B*n*c;
      gcn_sub_bf16<<<ceil256(tot), 256, 0, s>>>(x, xr, df, tot); }
    // pre = (x - x_r) @ tw^T + tb
    gemm(s, df, 0, c, 1, twb, 0, 1, c, pre, 0, c, B*n, c, c,
         tb_bias, 0, nullptr, 0, 1);
    gcn_bn_stats<<<c, 256, 0, s>>>(pre, B*n, c, mv, mv + c);
    { long tot = (long)B*n*c;
      gcn_bn_apply<<<ceil256(tot), 256, 0, s>>>(pre, mv, mv + c, g, be,
                                                x, x, xb, tot, c, 1); }
}

// direction conv block -> writes fm (f32) and fmb (bf16)
static void run_conv(hipStream_t s, Bump scr, const int* idx, const float* v, long vbs,
                     const __bf16* finb, int n, int cin, int cout,
                     const __bf16* wb, const float* bias, const float* dirsN,
                     const float* bng, const float* bnb,
                     float* fm, __bf16* fmb)
{
    const int B = 8;
    float* ff = (float*)scr.take((size_t)B*n*2*cout*4);
    float* co = (float*)scr.take((size_t)B*n*cout*4);
    float* mv = (float*)scr.take((size_t)cout*8);
    gemm(s, finb, 0, cin, 1, wb, 0, (long)(2*cout), 1, ff, 0, 2*cout,
         B*n, 2*cout, cin, bias, 0, nullptr, 0, 1);
    { dim3 g(n, B);
      gcn_conv_reduce<<<g, 128, 0, s>>>(idx, v, vbs, ff, dirsN, co, n, cout); }
    long tot = (long)B*n*cout;
    if (bng) {
        gcn_bn_stats<<<cout, 256, 0, s>>>(co, B*n, cout, mv, mv + cout);
        gcn_bn_apply<<<ceil256(tot), 256, 0, s>>>(co, mv, mv + cout, bng, bnb,
                                                  nullptr, fm, fmb, tot, cout, 1);
    } else {
        gcn_copy_cvt<<<ceil256(tot), 256, 0, s>>>(co, fm, fmb, tot);
    }
}

static inline void cvtw(hipStream_t s, const float* x, __bf16* y, long n)
{
    gcn_cvt_bf16<<<ceil256(n), 256, 0, s>>>(x, y, n);
}

// ------------------------------- launcher ----------------------------------
extern "C" void kernel_launch(void* const* d_in, const int* in_sizes, int n_in,
                              void* d_out, int out_size, void* d_ws, size_t ws_size,
                              hipStream_t stream)
{
    (void)in_sizes; (void)n_in; (void)out_size; (void)ws_size;
    const int B = 8, N0 = 2048, N1 = 512, N2 = 128, KNB = 50;
    const long VBS = (long)N0 * 3;

    const float* vertices = (const float*)d_in[0];
    const float* onehot   = (const float*)d_in[1];
    const float* conv0_dir= (const float*)d_in[2];
    const float* bn_g[4]  = {(const float*)d_in[3],  (const float*)d_in[15],
                             (const float*)d_in[27], (const float*)d_in[39]};
    const float* bn_b[4]  = {(const float*)d_in[4],  (const float*)d_in[16],
                             (const float*)d_in[28], (const float*)d_in[40]};
    // sa param base indices: sa0=5, sa1=17, sa2=29, sa3=41, sa4=51
    const int sab[5] = {5, 17, 29, 41, 51};
    const float* conv_w[4]   = {(const float*)d_in[12], (const float*)d_in[24],
                                (const float*)d_in[36], (const float*)d_in[48]};
    const float* conv_b[4]   = {(const float*)d_in[13], (const float*)d_in[25],
                                (const float*)d_in[37], (const float*)d_in[49]};
    const float* conv_dir[4] = {(const float*)d_in[14], (const float*)d_in[26],
                                (const float*)d_in[38], (const float*)d_in[50]};
    const float* h1_w = (const float*)d_in[58];
    const float* h1_b = (const float*)d_in[59];
    const float* h2_w = (const float*)d_in[60];
    const float* h2_b = (const float*)d_in[61];
    const float* h3_w = (const float*)d_in[62];
    const float* h3_b = (const float*)d_in[63];

    Bump ws{(char*)d_ws, 0};
    // --- persistent buffers ---
    float* v     = (float*)ws.take((size_t)B*N0*3*4);
    int*   idx0  = (int*)  ws.take((size_t)B*N0*KNB*4);
    int*   idx1  = (int*)  ws.take((size_t)B*N1*KNB*4);
    int*   idx2  = (int*)  ws.take((size_t)B*N2*KNB*4);
    float* fm0   = (float*)ws.take((size_t)B*N0*128*4);
    __bf16* fm0b = (__bf16*)ws.take((size_t)B*N0*128*2);
    float* fm1   = (float*)ws.take((size_t)B*N0*128*4);
    __bf16* fm1b = (__bf16*)ws.take((size_t)B*N0*128*2);
    float* fmp1  = (float*)ws.take((size_t)B*N1*128*4);
    __bf16* fmp1b= (__bf16*)ws.take((size_t)B*N1*128*2);
    float* fm2   = (float*)ws.take((size_t)B*N1*256*4);
    __bf16* fm2b = (__bf16*)ws.take((size_t)B*N1*256*2);
    float* fm3   = (float*)ws.take((size_t)B*N1*256*4);
    __bf16* fm3b = (__bf16*)ws.take((size_t)B*N1*256*2);
    float* fmp2  = (float*)ws.take((size_t)B*N2*256*4);
    __bf16* fmp2b= (__bf16*)ws.take((size_t)B*N2*256*2);
    float* fm4   = (float*)ws.take((size_t)B*N2*512*4);
    __bf16* fm4b = (__bf16*)ws.take((size_t)B*N2*512*2);
    float* fglob = (float*)ws.take((size_t)B*512*4);
    int*   ni1   = (int*)  ws.take((size_t)B*N0*4);
    int*   ni2   = (int*)  ws.take((size_t)B*N0*4);
    // normalized directions
    float* d0n = (float*)ws.take(3*128*4);
    float* dn[4];
    const int dirm[4] = {128, 256, 256, 512};
    for (int l = 0; l < 4; ++l) dn[l] = (float*)ws.take((size_t)3*dirm[l]*4);
    // bf16 weight copies
    const int cin_l[4]  = {128, 128, 256, 256};
    const int cout_l[4] = {128, 256, 256, 512};
    __bf16* cwb[4];
    for (int l = 0; l < 4; ++l)
        cwb[l] = (__bf16*)ws.take((size_t)cin_l[l] * 2 * cout_l[l] * 2);
    const int sac[5] = {128, 128, 256, 256, 512};
    __bf16 *qkb[5], *vwb[5], *twb[5];
    for (int l = 0; l < 5; ++l) {
        int c = sac[l];
        qkb[l] = (__bf16*)ws.take((size_t)(c/4)*c*2);
        vwb[l] = (__bf16*)ws.take((size_t)c*c*2);
        twb[l] = (__bf16*)ws.take((size_t)c*c*2);
    }
    __bf16* h1wb = (__bf16*)ws.take((size_t)1808*512*2);
    __bf16* h2wb = (__bf16*)ws.take((size_t)512*512*2);
    __bf16* h3wb = (__bf16*)ws.take((size_t)512*50*2);
    const size_t persistEnd = ws.off;

    // --- prep: transpose, weight conversion, dir normalization, knn ---
    { long tot = (long)B*3*N0;
      gcn_transpose_v<<<ceil256(tot), 256, 0, stream>>>(vertices, v, N0, tot); }
    for (int l = 0; l < 4; ++l)
        cvtw(stream, conv_w[l], cwb[l], (long)cin_l[l]*2*cout_l[l]);
    for (int l = 0; l < 5; ++l) {
        int c = sac[l];
        cvtw(stream, (const float*)d_in[sab[l]+0], qkb[l], (long)(c/4)*c);
        cvtw(stream, (const float*)d_in[sab[l]+1], vwb[l], (long)c*c);
        cvtw(stream, (const float*)d_in[sab[l]+3], twb[l], (long)c*c);
    }
    cvtw(stream, h1_w, h1wb, (long)1808*512);
    cvtw(stream, h2_w, h2wb, (long)512*512);
    cvtw(stream, h3_w, h3wb, (long)512*50);
    gcn_norm_dirs<<<1, 128, 0, stream>>>(conv0_dir, d0n, 128);
    for (int l = 0; l < 4; ++l)
        gcn_norm_dirs<<<(dirm[l] + 255)/256, 256, 0, stream>>>(
            conv_dir[l], dn[l], dirm[l]);
    gcn_knn<<<dim3(N0, B), 32, 0, stream>>>(v, VBS, N0, idx0, KNB);

    // --- conv_surface + bn0 -> fm0, then sa0 ---
    {
        Bump scr{(char*)d_ws, persistEnd};
        float* co = (float*)scr.take((size_t)B*N0*128*4);
        float* mv = (float*)scr.take(128*8);
        gcn_conv_surface<<<dim3(N0, B), 128, 0, stream>>>(idx0, v, VBS, d0n, co, N0, 128);
        gcn_bn_stats<<<128, 256, 0, stream>>>(co, B*N0, 128, mv, mv + 128);
        long tot = (long)B*N0*128;
        gcn_bn_apply<<<ceil256(tot), 256, 0, stream>>>(co, mv, mv + 128,
            bn_g[0], bn_b[0], nullptr, fm0, fm0b, tot, 128, 1);
    }
    { Bump scr{(char*)d_ws, persistEnd};
      run_sa(stream, scr, fm0, fm0b, N0, 128, qkb[0], vwb[0],
             (const float*)d_in[sab[0]+2], twb[0], (const float*)d_in[sab[0]+4],
             (const float*)d_in[sab[0]+5], (const float*)d_in[sab[0]+6]); }

    // --- conv1 + bn1 -> fm1, sa1 ---
    { Bump scr{(char*)d_ws, persistEnd};
      run_conv(stream, scr, idx0, v, VBS, fm0b, N0, 128, 128,
               cwb[0], conv_b[0], dn[0], bn_g[1], bn_b[1], fm1, fm1b); }
    { Bump scr{(char*)d_ws, persistEnd};
      run_sa(stream, scr, fm1, fm1b, N0, 128, qkb[1], vwb[1],
             (const float*)d_in[sab[1]+2], twb[1], (const float*)d_in[sab[1]+4],
             (const float*)d_in[sab[1]+5], (const float*)d_in[sab[1]+6]); }

    // --- pool1 + knn1 ---
    gcn_pool<<<dim3(N1, B), 32, 0, stream>>>(v, VBS, N0, fm1, 128, N1, fmp1, fmp1b);
    gcn_knn<<<dim3(N1, B), 32, 0, stream>>>(v, VBS, N1, idx1, KNB);

    // --- conv2 + bn2 -> fm2, sa2 ---
    { Bump scr{(char*)d_ws, persistEnd};
      run_conv(stream, scr, idx1, v, VBS, fmp1b, N1, 128, 256,
               cwb[1], conv_b[1], dn[1], bn_g[2], bn_b[2], fm2, fm2b); }
    { Bump scr{(char*)d_ws, persistEnd};
      run_sa(stream, scr, fm2, fm2b, N1, 256, qkb[2], vwb[2],
             (const float*)d_in[sab[2]+2], twb[2], (const float*)d_in[sab[2]+4],
             (const float*)d_in[sab[2]+5], (const float*)d_in[sab[2]+6]); }

    // --- conv3 + bn3 -> fm3, sa3 ---
    { Bump scr{(char*)d_ws, persistEnd};
      run_conv(stream, scr, idx1, v, VBS, fm2b, N1, 256, 256,
               cwb[2], conv_b[2], dn[2], bn_g[3], bn_b[3], fm3, fm3b); }
    { Bump scr{(char*)d_ws, persistEnd};
      run_sa(stream, scr, fm3, fm3b, N1, 256, qkb[3], vwb[3],
             (const float*)d_in[sab[3]+2], twb[3], (const float*)d_in[sab[3]+4],
             (const float*)d_in[sab[3]+5], (const float*)d_in[sab[3]+6]); }

    // --- pool2 + knn2 ---
    gcn_pool<<<dim3(N2, B), 32, 0, stream>>>(v, VBS, N1, fm3, 256, N2, fmp2, fmp2b);
    gcn_knn<<<dim3(N2, B), 32, 0, stream>>>(v, VBS, N2, idx2, KNB);

    // --- conv4 (no bn) -> fm4, sa4 ---
    { Bump scr{(char*)d_ws, persistEnd};
      run_conv(stream, scr, idx2, v, VBS, fmp2b, N2, 256, 512,
               cwb[3], conv_b[3], dn[3], nullptr, nullptr, fm4, fm4b); }
    { Bump scr{(char*)d_ws, persistEnd};
      run_sa(stream, scr, fm4, fm4b, N2, 512, qkb[4], vwb[4],
             (const float*)d_in[sab[4]+2], twb[4], (const float*)d_in[sab[4]+4],
             (const float*)d_in[sab[4]+5], (const float*)d_in[sab[4]+6]); }

    // --- head: global max, nearest upsample, fuse, MLP, log-softmax ---
    { dim3 g((512+255)/256, B); gcn_rowmax<<<g, 256, 0, stream>>>(fm4, N2, 512, fglob); }
    gcn_nearest<<<dim3(N0, B), 32, 0, stream>>>(v, VBS, N1, ni1, N0);
    gcn_nearest<<<dim3(N0, B), 32, 0, stream>>>(v, VBS, N2, ni2, N0);

    Bump scr{(char*)d_ws, persistEnd};
    __bf16* fuseb  = (__bf16*)scr.take((size_t)B*N0*1808*2);
    float*  x1     = (float*) scr.take((size_t)B*N0*512*4);
    __bf16* x1b    = (__bf16*)scr.take((size_t)B*N0*512*2);
    float*  x2     = (float*) scr.take((size_t)B*N0*512*4);
    __bf16* x2b    = (__bf16*)scr.take((size_t)B*N0*512*2);
    float*  logits = (float*) scr.take((size_t)B*N0*50*4);

    gcn_fuse<<<dim3(N0, B), 256, 0, stream>>>(fm0, fm1, fm2, fm3, fm4, fglob,
                                              onehot, ni1, ni2, fuseb, N0);
    gemm(stream, fuseb, 0, 1808, 1, h1wb, 0, 512, 1, x1, 0, 512,
         B*N0, 512, 1808, h1_b, 1, x1b, 0, 1);
    gemm(stream, x1b, 0, 512, 1, h2wb, 0, 512, 1, x2, 0, 512,
         B*N0, 512, 512, h2_b, 1, x2b, 0, 1);
    gemm(stream, x2b, 0, 512, 1, h3wb, 0, 50, 1, logits, 0, 50,
         B*N0, 50, 512, h3_b, 0, nullptr, 0, 1);
    gcn_logsoftmax<<<B*N0, 64, 0, stream>>>(logits, (float*)d_out, 50);
}
